// CustomLSTM_65575560675736
// MI455X (gfx1250) — matrix operand order repaired and checked by reading, not compiled, not run on hardware
//
#include <hip/hip_runtime.h>
#include <cstdint>

// ---------------------------------------------------------------------------
// Fused LSTM for MI455X (gfx1250, wave32, WMMA).
//   gates_t = [x_t | h_{t-1}] (64x384, bf16) @ Wt^T (384x1024, bf16) + bias
// 16 workgroups; WG g owns hidden columns [16g,16g+16) across ALL 4 gates so
// the c/h elementwise update is WG-local. 8 waves/WG: wave = (K-part 0..3) x
// (M-half 0..1); each wave's weight slice (96 VGPRs of B-fragments) is loop-
// invariant but is re-loaded from global Wt every timestep. Partials
// reduced in LDS in a fixed (deterministic) order. Device-wide step sync via
// a sense-reversing atomic barrier in workspace.
// ---------------------------------------------------------------------------

typedef __attribute__((ext_vector_type(16))) __bf16   v16bf;
typedef __attribute__((ext_vector_type(8)))  float    v8f;
typedef __attribute__((ext_vector_type(4)))  unsigned u32x4;

union FragAB { u32x4 u[2]; v16bf v; };

#define LSTM_B  64
#define LSTM_T  1024
#define LSTM_I  128
#define LSTM_H  256
#define LSTM_K  384          // I + H (fused)
#define LSTM_G4 1024         // 4*H
#define NWG     16
#define BLOCK   256          // 8 waves: (K-part 4) x (M-half 2)

__device__ __forceinline__ unsigned short f2bf(float f) {
  unsigned u = __float_as_uint(f);
  unsigned r = ((u >> 16) & 1u) + 0x7FFFu;     // round-to-nearest-even
  return (unsigned short)((u + r) >> 16);
}

__device__ __forceinline__ float sigf(float x) {
  return 1.0f / (1.0f + __expf(-x));
}

// x (B,T,I) f32 -> bf16 (same layout)
__global__ void k_cvt_x(const float* __restrict__ x,
                        unsigned short* __restrict__ xb, int n) {
  int i = blockIdx.x * blockDim.x + threadIdx.x;
  if (i < n) xb[i] = f2bf(x[i]);
}

// Build Wt[c][k] (c=gate column 0..1023, k=0..383): k<128 -> W[k][c], else U[k-128][c]
__global__ void k_build_w(const float* __restrict__ W,
                          const float* __restrict__ U,
                          unsigned short* __restrict__ Wt, int n) {
  int i = blockIdx.x * blockDim.x + threadIdx.x;
  if (i >= n) return;
  int c = i / LSTM_K;
  int k = i - c * LSTM_K;
  float v = (k < LSTM_I) ? W[(size_t)k * LSTM_G4 + c]
                         : U[(size_t)(k - LSTM_I) * LSTM_G4 + c];
  Wt[i] = f2bf(v);
}

__global__ void k_zero(unsigned* __restrict__ p, int n) {
  int i = blockIdx.x * blockDim.x + threadIdx.x;
  if (i < n) p[i] = 0u;
}

__global__ __launch_bounds__(BLOCK) void k_lstm(
    const unsigned short* __restrict__ xb,   // [B][T][I] bf16
    const unsigned short* __restrict__ Wt,   // [4H][K]   bf16
    unsigned short* __restrict__ hbuf,       // [2][B][H] bf16 (double buffer)
    const float* __restrict__ bias,          // [4H]
    float* __restrict__ hfinal,              // [B][H] f32
    unsigned* __restrict__ bar)              // [cnt, gen]
{
  __shared__ float part[4][4][4][256];       // [kpart][gate][mtile][16x16] = 64KB
  __shared__ float cst[LSTM_B][16];          // cell state for this WG's columns

  const int tid  = threadIdx.x;
  const int wv   = tid >> 5;                 // wave 0..7
  const int kp   = wv & 3;                   // K partition: [kp*96, kp*96+96)
  const int mh   = wv >> 2;                  // M half: tiles {2*mh, 2*mh+1}
  const int lane = tid & 31;
  const int hi   = lane >> 4;                // half-wave 0/1
  const int ln   = lane & 15;
  const int wg   = blockIdx.x;
  const int col0 = wg * 16;                  // owned hidden columns

  for (int e = tid; e < LSTM_B * 16; e += BLOCK) cst[e >> 4][e & 15] = 0.0f;
  __syncthreads();

  // Phase-B per-thread constants (bias hoisted out of the serial t-loop).
  const int nB = tid & 15;                   // hidden column (local)
  const int mB = tid >> 4;                   // base batch row, +16 per iter
  float bs[4];
  #pragma unroll
  for (int gi = 0; gi < 4; ++gi) bs[gi] = bias[gi * LSTM_H + col0 + nB];

  unsigned* cnt = bar;
  unsigned* gen = bar + 1;

  for (int t = 0; t < LSTM_T; ++t) {
    const unsigned short* hread = hbuf + (size_t)(t & 1) * (LSTM_B * LSTM_H);
    unsigned short* hwrite      = hbuf + (size_t)((t + 1) & 1) * (LSTM_B * LSTM_H);

    // ---- phase A: partial gate GEMM via WMMA; wave covers K in [kp*96, kp*96+96)
    for (int gi = 0; gi < 4; ++gi) {
      const int gcol = gi * LSTM_H + col0 + ln;          // weight row (gate column)
      #pragma unroll
      for (int mti = 0; mti < 2; ++mti) {
        const int mt = mh * 2 + mti;
        const int m  = mt * 16 + ln;                     // batch row
        v8f acc = {};
        #pragma unroll
        for (int kk = 0; kk < 3; ++kk) {
          const int k0 = (kp * 3 + kk) * 32;
          // A fragment (16x32 bf16): per-lane 8 elems at k0 + 8*hi, next 8 at +16
          const unsigned short* abase =
              (k0 < LSTM_I)
                  ? xb + ((size_t)m * LSTM_T + t) * LSTM_I + k0 + hi * 8
                  : hread + (size_t)m * LSTM_H + (k0 - LSTM_I) + hi * 8;
          FragAB fa;
          const u32x4* ap = (const u32x4*)abase;
          fa.u[0] = ap[0];
          fa.u[1] = ap[2];                               // +16 elems = +32B
          // B fragment (32x16 bf16): per-lane column gcol, 8 elems at k0 + 8*hi, next 8 at +16
          const unsigned short* bbase =
              Wt + (size_t)gcol * LSTM_K + k0 + hi * 8;
          const u32x4* bp = (const u32x4*)bbase;
          FragAB fb;
          fb.u[0] = bp[0];
          fb.u[1] = bp[2];
          acc = __builtin_amdgcn_wmma_f32_16x16x32_bf16(
              false, fa.v, false, fb.v, (short)0, acc, false, false);
        }
        // D layout: VGPR r -> M = 8*hi + r, N = ln
        float* pp = &part[kp][gi][mt][0];
        #pragma unroll
        for (int r = 0; r < 8; ++r) pp[(hi * 8 + r) * 16 + ln] = acc[r];
      }
    }
    __syncthreads();

    // ---- phase B: deterministic reduction + bias + activations + state update
    // 1024 elems / 256 threads = 4 iters; iter ii handles M-tile mt == ii.
    #pragma unroll
    for (int ii = 0; ii < 4; ++ii) {
      const int m   = mB + ii * 16;
      const int idx = mB * 16 + nB;          // (m & 15) * 16 + n
      float g[4];
      #pragma unroll
      for (int gi = 0; gi < 4; ++gi) {
        g[gi] = ((part[0][gi][ii][idx] + part[1][gi][ii][idx]) +
                 (part[2][gi][ii][idx] + part[3][gi][ii][idx])) +
                bs[gi];
      }
      const float it = sigf(g[0]);
      const float ft = sigf(g[1]);
      const float gt = tanhf(g[2]);
      const float ot = sigf(g[3]);
      const float c  = ft * cst[m][nB] + it * gt;
      cst[m][nB] = c;
      const float h = ot * tanhf(c);
      hwrite[(size_t)m * LSTM_H + col0 + nB] = f2bf(h);
      if (t == LSTM_T - 1) hfinal[(size_t)m * LSTM_H + col0 + nB] = h;
    }

    // ---- phase C: device-wide barrier across the 16 resident workgroups
    __threadfence();
    __syncthreads();
    if (tid == 0) {
      const unsigned g = *(volatile unsigned*)gen;
      if (atomicAdd(cnt, 1u) + 1u == (unsigned)NWG) {
        *(volatile unsigned*)cnt = 0u;
        __threadfence();
        *(volatile unsigned*)gen = g + 1u;
      } else {
        while (*(volatile unsigned*)gen == g) __builtin_amdgcn_s_sleep(2);
      }
    }
    __syncthreads();
    __threadfence();
  }
}

// out[b][o] = hfinal[b] . dense_w[o] + dense_b[o]
__global__ void k_dense(const float* __restrict__ hf,
                        const float* __restrict__ dw,
                        const float* __restrict__ db,
                        float* __restrict__ out) {
  int tid = blockIdx.x * blockDim.x + threadIdx.x;
  if (tid >= LSTM_B * 128) return;
  int b = tid >> 7, o = tid & 127;
  const float* hp = hf + (size_t)b * LSTM_H;
  const float* wp = dw + (size_t)o * LSTM_H;
  float acc = db[o];
  for (int k = 0; k < LSTM_H; ++k) acc += hp[k] * wp[k];
  out[tid] = acc;
}

extern "C" void kernel_launch(void* const* d_in, const int* in_sizes, int n_in,
                              void* d_out, int out_size, void* d_ws, size_t ws_size,
                              hipStream_t stream) {
  const float* x    = (const float*)d_in[0];
  const float* W    = (const float*)d_in[1];
  const float* U    = (const float*)d_in[2];
  const float* bias = (const float*)d_in[3];
  const float* dw   = (const float*)d_in[4];
  const float* db   = (const float*)d_in[5];
  float* out = (float*)d_out;

  char* ws = (char*)d_ws;
  size_t off = 0;
  auto carve = [&](size_t bytes) {
    char* p = ws + off;
    off = (off + bytes + 255) & ~(size_t)255;
    return p;
  };
  unsigned short* xb = (unsigned short*)carve((size_t)LSTM_B * LSTM_T * LSTM_I * 2); // 16 MB
  unsigned short* Wt = (unsigned short*)carve((size_t)LSTM_G4 * LSTM_K * 2);         // 768 KB
  unsigned short* hb = (unsigned short*)carve((size_t)2 * LSTM_B * LSTM_H * 2);      // 64 KB
  unsigned*       bar= (unsigned*)carve(256);                                        // contiguous after hb
  float*          hf = (float*)carve((size_t)LSTM_B * LSTM_H * 4);

  const int nx = LSTM_B * LSTM_T * LSTM_I;            // 8388608
  const int nw = LSTM_G4 * LSTM_K;                    // 393216
  const int nz = (2 * LSTM_B * LSTM_H * 2 + 256) / 4; // hb + barrier, as u32

  k_cvt_x  <<<(nx + 255) / 256, 256, 0, stream>>>(x, xb, nx);
  k_build_w<<<(nw + 255) / 256, 256, 0, stream>>>(W, U, Wt, nw);
  k_zero   <<<(nz + 255) / 256, 256, 0, stream>>>((unsigned*)hb, nz);
  k_lstm   <<<NWG, BLOCK, 0, stream>>>(xb, Wt, hb, bias, hf, bar);
  k_dense  <<<(LSTM_B * 128 + 255) / 256, 256, 0, stream>>>(hf, dw, db, out);
}
